// HAN_20323785245264
// MI455X (gfx1250) — matrix operand order repaired and checked by reading, not compiled, not run on hardware
//
#include <hip/hip_runtime.h>
#include <hip/hip_bf16.h>
#include <math.h>

typedef __attribute__((ext_vector_type(16))) __bf16 v16bf;
typedef __attribute__((ext_vector_type(8)))  float  v8f;

#define DEVFN __device__ __forceinline__

// monotone float <-> uint encoding so atomicMax(u32) == float max
DEVFN unsigned fenc(float f){ unsigned u=__float_as_uint(f); return (u&0x80000000u)?~u:(u|0x80000000u); }
DEVFN float    fdec(unsigned u){ return __uint_as_float((u&0x80000000u)?(u^0x80000000u):~u); }

// convert fp32 -> bf16 (RNE via v_cvt), used once per call on the small weight mats
__global__ void cvt_bf16(const float* __restrict__ src, __bf16* __restrict__ dst, int n)
{
  const int t = blockIdx.x * blockDim.x + threadIdx.x;
  if (t < n) dst[t] = (__bf16)src[t];
}

// ---------------------------------------------------------------------------
// WMMA GEMM: Y[N,64] = X[N,64] @ W[64,64]^T + bias (X fp32->bf16 inline,
// W pre-converted bf16, f32 accumulate).
// One wave handles 16 rows x 64 outputs: 4 N-tiles x 2 K-steps = 8 WMMAs.
// A layout (this code, contiguous k): lanes 0-15 -> K 0..15, lanes 16-31 -> K 16..31.
// B layout (this code, same contiguous k as A): lanes 0-15 -> K 0..15, lanes 16-31 -> K 16..31.
// C/D layout (32-bit 16x16): vgpr r -> M = r + (lane<16 ? 0 : 8), N = lane%16.
// ---------------------------------------------------------------------------
__global__ void gemm64_bias(const float* __restrict__ X, const __bf16* __restrict__ Wbf,
                            const float* __restrict__ B, float* __restrict__ Y, int N)
{
  const int wave    = threadIdx.x >> 5;
  const int lane    = threadIdx.x & 31;
  const int rowBase = (blockIdx.x * 8 + wave) * 16;
  if (rowBase >= N) return;                       // wave-uniform; EXEC stays full below
  const int lm  = lane & 15;
  const int hi  = lane >> 4;                      // 0 or 1
  const int kbA = hi * 16;
  int r = rowBase + lm; if (r > N - 1) r = N - 1; // clamp loads, mask ragged stores
  const float* xr = X + (size_t)r * 64;
  v16bf a0, a1;
#pragma unroll
  for (int i = 0; i < 16; ++i) {
    const int k = kbA + i;
    a0[i] = (__bf16)xr[k];
    a1[i] = (__bf16)xr[32 + k];
  }
  const int kbB = hi * 16;
  const bool fullTile = (rowBase + 16 <= N);      // wave-uniform
#pragma unroll
  for (int nt = 0; nt < 4; ++nt) {
    const int n = nt * 16 + lm;
    // 16 contiguous bf16 (32B, 32B-aligned) per K-chunk: direct vector loads
    const v16bf b0 = *(const v16bf*)(Wbf + (size_t)n * 64 + kbB);
    const v16bf b1 = *(const v16bf*)(Wbf + (size_t)n * 64 + 32 + kbB);
    v8f c = {};
    c = __builtin_amdgcn_wmma_f32_16x16x32_bf16(false, a0, false, b0, (short)0, c, false, false);
    c = __builtin_amdgcn_wmma_f32_16x16x32_bf16(false, a1, false, b1, (short)0, c, false, false);
    const float bv = B[n];
    float* yp = Y + (size_t)(rowBase + hi * 8) * 64 + n;
    if (fullTile) {
#pragma unroll
      for (int rr = 0; rr < 8; ++rr) yp[(size_t)rr * 64] = c[rr] + bv;
    } else {
#pragma unroll
      for (int rr = 0; rr < 8; ++rr) {
        const int row = rowBase + rr + hi * 8;
        if (row < N) Y[(size_t)row * 64 + n] = c[rr] + bv;
      }
    }
  }
}

// per-head attention dot:  out[n,h] = sum_d H[n, h*d+j] * att[h*d+j]
__global__ void att_dot(const float* __restrict__ H, const float* __restrict__ att,
                        float* __restrict__ out, int N, int heads)
{
  const int t = blockIdx.x * blockDim.x + threadIdx.x;
  if (t >= N * heads) return;
  const int n = t / heads, h = t - n * heads;
  const int d = 64 / heads;
  const float* hp = H + (size_t)n * 64 + h * d;
  const float* ap = att + h * d;
  float s = 0.f;
  for (int j = 0; j < d; ++j) s += hp[j] * ap[j];
  out[t] = s;
}

__global__ void edge_max(const float* __restrict__ as, const float* __restrict__ ad,
                         const int* __restrict__ si, const int* __restrict__ di,
                         unsigned* __restrict__ menc, int E, int heads)
{
  const int t = blockIdx.x * blockDim.x + threadIdx.x;
  if (t >= E * heads) return;
  const int e = t / heads, h = t - e * heads;
  float l = as[(size_t)si[e] * heads + h] + ad[(size_t)di[e] * heads + h];
  l = (l > 0.f) ? l : 0.1f * l;                   // leaky_relu, NEG_SLOPE=0.1
  atomicMax(&menc[(size_t)di[e] * heads + h], fenc(l));
}

__global__ void edge_sum(const float* __restrict__ as, const float* __restrict__ ad,
                         const int* __restrict__ si, const int* __restrict__ di,
                         const unsigned* __restrict__ menc, float* __restrict__ ssum,
                         int E, int heads)
{
  const int t = blockIdx.x * blockDim.x + threadIdx.x;
  if (t >= E * heads) return;
  const int e = t / heads, h = t - e * heads;
  const int d = di[e];
  float l = as[(size_t)si[e] * heads + h] + ad[(size_t)d * heads + h];
  l = (l > 0.f) ? l : 0.1f * l;
  const float m = fdec(menc[(size_t)d * heads + h]);
  atomicAdd(&ssum[(size_t)d * heads + h], __expf(l - m));
}

__global__ void edge_scatter(const float* __restrict__ Hsrc,
                             const float* __restrict__ as, const float* __restrict__ ad,
                             const int* __restrict__ si, const int* __restrict__ di,
                             const unsigned* __restrict__ menc, const float* __restrict__ ssum,
                             float* __restrict__ agg, int E, int heads)
{
  const int t = blockIdx.x * blockDim.x + threadIdx.x;
  if (t >= E * heads) return;
  const int e = t / heads, h = t - e * heads;
  const int s = si[e], d = di[e];
  float l = as[(size_t)s * heads + h] + ad[(size_t)d * heads + h];
  l = (l > 0.f) ? l : 0.1f * l;
  const float m     = fdec(menc[(size_t)d * heads + h]);
  const float alpha = __expf(l - m) / (ssum[(size_t)d * heads + h] + 1e-16f);
  const int dd = 64 / heads;
  const float* hp = Hsrc + (size_t)s * 64 + h * dd;
  float*       ap = agg  + (size_t)d * 64 + h * dd;
  for (int j = 0; j < dd; ++j) atomicAdd(&ap[j], alpha * hp[j]);
}

__global__ void relu_ip(float* __restrict__ x, size_t n)
{
  const size_t t = (size_t)blockIdx.x * blockDim.x + threadIdx.x;
  if (t < n) x[t] = fmaxf(x[t], 0.f);
}

// column sums of tanh(T) over rows -> cs[64]  (for semantic attention mean)
__global__ void tanh_colsum(const float* __restrict__ T, float* __restrict__ cs, int N)
{
  __shared__ float sh[64];
  const int t = threadIdx.x;
  if (t < 64) sh[t] = 0.f;
  __syncthreads();
  const int col = t & 63;
  const int rpb = blockDim.x >> 6;
  float acc = 0.f;
  for (int row = blockIdx.x * rpb + (t >> 6); row < N; row += gridDim.x * rpb)
    acc += tanhf(T[(size_t)row * 64 + col]);
  atomicAdd(&sh[col], acc);
  __syncthreads();
  if (t < 64) atomicAdd(&cs[t], sh[t]);
}

__global__ void score_softmax(const float* __restrict__ cs, const float* __restrict__ q,
                              float* __restrict__ attn, int M, float invN)
{
  if (threadIdx.x != 0 || blockIdx.x != 0) return;
  float sc[8]; float mx = -1e30f;
  for (int m = 0; m < M; ++m) {
    float s = 0.f;
    for (int f = 0; f < 64; ++f) s += q[f] * cs[m * 64 + f];
    s *= invN; sc[m] = s; if (s > mx) mx = s;
  }
  float tot = 0.f;
  for (int m = 0; m < M; ++m) { sc[m] = __expf(sc[m] - mx); tot += sc[m]; }
  for (int m = 0; m < M; ++m) attn[m] = sc[m] / tot;
}

struct Ptr5 { const float* p[5]; };

__global__ void combine(Ptr5 aggs, const float* __restrict__ attn,
                        float* __restrict__ res, int M, size_t n64)
{
  const size_t t = (size_t)blockIdx.x * blockDim.x + threadIdx.x;
  if (t >= n64) return;
  float s = 0.f;
  for (int m = 0; m < M; ++m) s += attn[m] * aggs.p[m][t];
  res[t] = s;
}

// in-place ELU (+ optional LayerNorm) per 64-wide row, thread-per-row
__global__ void elu_ln(float* __restrict__ x, const float* __restrict__ g,
                       const float* __restrict__ b, int N, int doLN)
{
  const int n = blockIdx.x * blockDim.x + threadIdx.x;
  if (n >= N) return;
  float* r = x + (size_t)n * 64;
  float mu = 0.f;
  for (int f = 0; f < 64; ++f) {
    float v = r[f];
    v = (v > 0.f) ? v : expm1f(v);
    r[f] = v; mu += v;
  }
  if (!doLN) return;
  mu *= (1.f / 64.f);
  float var = 0.f;
  for (int f = 0; f < 64; ++f) { const float d = r[f] - mu; var += d * d; }
  var *= (1.f / 64.f);
  const float is = rsqrtf(var + 1e-5f);
  for (int f = 0; f < 64; ++f) r[f] = (r[f] - mu) * is * g[f] + b[f];
}

// flash-style set-transformer attention: block = one (head,probe), online softmax over N
__global__ void st_attn(const float* __restrict__ Q, const float* __restrict__ K,
                        const float* __restrict__ V, float* __restrict__ O, int N)
{
  const int h = blockIdx.x >> 4;
  const int p = blockIdx.x & 15;
  const int t = threadIdx.x;
  float q[16];
#pragma unroll
  for (int j = 0; j < 16; ++j) q[j] = Q[p * 64 + h * 16 + j] * 0.25f;   // 1/sqrt(d=16)
  float m = -1e30f, l = 0.f, acc[16];
#pragma unroll
  for (int j = 0; j < 16; ++j) acc[j] = 0.f;
  for (int n = t; n < N; n += 256) {
    const float* kr = K + (size_t)n * 64 + h * 16;
    float s = 0.f;
#pragma unroll
    for (int j = 0; j < 16; ++j) s += q[j] * kr[j];
    const float nm = fmaxf(m, s);
    const float co = __expf(m - nm);
    const float e  = __expf(s - nm);
    l = l * co + e;
    const float* vr = V + (size_t)n * 64 + h * 16;
#pragma unroll
    for (int j = 0; j < 16; ++j) acc[j] = acc[j] * co + e * vr[j];
    m = nm;
  }
  __shared__ float sm[256], sl[256], sa[256 * 17];  // stride 17: no bank conflicts
  sm[t] = m; sl[t] = l;
#pragma unroll
  for (int j = 0; j < 16; ++j) sa[t * 17 + j] = acc[j];
  __syncthreads();
  for (int off = 128; off > 0; off >>= 1) {
    if (t < off) {
      const float m1 = sm[t], m2 = sm[t + off];
      const float M  = fmaxf(m1, m2);
      const float c1 = __expf(m1 - M), c2 = __expf(m2 - M);
      sl[t] = sl[t] * c1 + sl[t + off] * c2;
      for (int j = 0; j < 16; ++j)
        sa[t * 17 + j] = sa[t * 17 + j] * c1 + sa[(t + off) * 17 + j] * c2;
      sm[t] = M;
    }
    __syncthreads();
  }
  if (t < 16) O[p * 64 + h * 16 + t] = sa[t] / sl[0];
}

// omean[f] = mean_p ( O[p,:] @ W[f,:] + b[f] )
__global__ void st_mean(const float* __restrict__ O, const float* __restrict__ W,
                        const float* __restrict__ B, float* __restrict__ omean)
{
  const int f = threadIdx.x;
  if (f >= 64) return;
  float s = 0.f;
  for (int p = 0; p < 16; ++p) {
    float t = B[f];
    for (int k = 0; k < 64; ++k) t += O[p * 64 + k] * W[f * 64 + k];
    s += t;
  }
  omean[f] = s * (1.f / 16.f);
}

// tiny matvec head: act 0=none, 1=gelu(exact), 2=leaky(0.01), 3=relu
__global__ void small_linear(const float* __restrict__ x, const float* __restrict__ W,
                             const float* __restrict__ B, float* __restrict__ y,
                             int n_out, int n_in, int act)
{
  const int o = threadIdx.x;
  if (o >= n_out) return;
  float s = B[o];
  for (int k = 0; k < n_in; ++k) s += x[k] * W[o * n_in + k];
  if (act == 1)      s = 0.5f * s * (1.f + erff(s * 0.70710678118654752f));
  else if (act == 2) s = (s > 0.f) ? s : 0.01f * s;
  else if (act == 3) s = fmaxf(s, 0.f);
  y[o] = s;
}

// ---------------------------------------------------------------------------
extern "C" void kernel_launch(void* const* d_in, const int* in_sizes, int n_in,
                              void* d_out, int out_size, void* d_ws, size_t ws_size,
                              hipStream_t stream)
{
  (void)in_sizes; (void)n_in; (void)out_size; (void)ws_size;
  auto F = [&](int i) { return (const float*)d_in[i]; };
  auto I = [&](int i) { return (const int*)d_in[i]; };
  auto cdiv = [](long a, long b) { return (int)((a + b - 1) / b); };

  static const int NNh[3] = {100000, 50000, 20000};     // inst, var, const
  struct ETT { int s, d, E, si, di; };
  static const ETT ets[7] = {                           // EDGES insertion order
    {0,0,800000, 3, 4},   // cc
    {0,0,200000, 5, 6},   // call
    {0,1,600000, 7, 8},   // iv
    {1,0,600000, 9,10},   // vi
    {2,0,300000,11,12},   // ci
    {0,0,200000,13,14},   // ii
    {1,1,200000,15,16}};  // vv

  // ----- workspace layout (floats) -----
  float* base = (float*)d_ws;
  size_t off = 0;
  auto alloc = [&](size_t n) { float* p = base + off; off += n; return p; };
  float* H[3]; for (int t = 0; t < 3; ++t) H[t] = alloc((size_t)NNh[t] * 64);
  float *as[7], *ad[7], *me[7], *ss[7], *agg[7];
  for (int e = 0; e < 7; ++e) {
    as[e] = alloc((size_t)NNh[ets[e].s] * 8);
    ad[e] = alloc((size_t)NNh[ets[e].d] * 8);
    me[e] = alloc((size_t)NNh[ets[e].d] * 8);
    ss[e] = alloc((size_t)NNh[ets[e].d] * 8);
  }
  for (int e = 0; e < 7; ++e) agg[e] = alloc((size_t)NNh[ets[e].d] * 64);
  float* T      = alloc((size_t)100000 * 64);
  float* colsum = alloc(8 * 64);
  float* attnb  = alloc(16);
  float* res[2]; res[0] = alloc((size_t)100000 * 64); res[1] = alloc((size_t)50000 * 64);
  float* Qb = alloc(16 * 64); float* Ob = alloc(16 * 64); float* om = alloc(64);
  float* stout = alloc(128);  float* gb = alloc(128);     float* g2 = alloc(64);

  // ----- bf16 weight arena: convert all WMMA-consumed weights once per call -----
  // proj W: 17,19,21 (L1), 40,42 (L2), 61,63 (L3); semantic k.w: 38,59,80  -> 64x64 each
  // set-transformer in_w (192x64): 86, 93
  __bf16* wbf[108] = {};                                  // index by param leaf id
  {
    static const int w64[10] = {17,19,21,40,42,61,63,38,59,80};
    for (int i = 0; i < 10; ++i) {
      __bf16* dst = (__bf16*)alloc(2048);                 // 4096 bf16 = 2048 floats
      wbf[w64[i]] = dst;
      cvt_bf16<<<cdiv(4096,256),256,0,stream>>>(F(w64[i]), dst, 4096);
    }
    static const int w192[2] = {86, 93};
    for (int i = 0; i < 2; ++i) {
      __bf16* dst = (__bf16*)alloc(6144);                 // 12288 bf16
      wbf[w192[i]] = dst;
      cvt_bf16<<<cdiv(12288,256),256,0,stream>>>(F(w192[i]), dst, 12288);
    }
  }

  // ----- param leaf indices (setup_inputs insertion order, recursive) -----
  // 0..2 node feats; 3..16 edge idx; 17.. params:
  // han1: 17/18 projI, 19/20 projV, 21/22 projC, 23..36 att pairs, 37 q, 38/39 k
  // han2: 40/41 projI, 42/43 projV, 44..57 att, 58 q, 59/60 k
  // han3: 61/62 projI, 63/64 projV, 65..78 att, 79 q, 80/81 k
  // 82/83 norm1, 84/85 norm2
  // st_inst 86..92: in_w,in_b,out_w,out_b,ind,fc_w,fc_b ; st_var 93..99
  // 100/101 fc_inst, 102/103 fc_var, 104/105 fc_graph, 106/107 fc_out
  struct LP { int pw[3], pb[3], attB, qi, kwi, kbi, heads; bool ci, cst, ln; int lng, lnb; };
  static const LP L[3] = {
    {{17,19,21}, {18,20,22}, 23, 37, 38, 39, 8, true,  true,  true, 82, 83},
    {{40,42,-1}, {41,43,-1}, 44, 58, 59, 60, 8, false, false, true, 84, 85},
    {{61,63,-1}, {62,64,-1}, 65, 79, 80, 81, 4, false, false, false, 0,  0}};

  for (int li = 0; li < 3; ++li) {
    const LP& lp = L[li];
    const int heads = lp.heads;
    const float* xin[3] = { li == 0 ? F(0) : res[0], li == 0 ? F(1) : res[1],
                            li == 0 ? F(2) : (const float*)nullptr };
    const int ntypes = lp.cst ? 3 : 2;
    for (int t = 0; t < ntypes; ++t)
      gemm64_bias<<<cdiv(NNh[t],128),256,0,stream>>>(xin[t], wbf[lp.pw[t]], F(lp.pb[t]), H[t], NNh[t]);

    for (int e = 0; e < 7; ++e) {
      if (!lp.ci && e == 4) continue;               // 'ci' dropped after layer 1
      const ETT& et = ets[e];
      att_dot<<<cdiv((long)NNh[et.s]*heads,256),256,0,stream>>>(H[et.s], F(lp.attB+2*e),   as[e], NNh[et.s], heads);
      att_dot<<<cdiv((long)NNh[et.d]*heads,256),256,0,stream>>>(H[et.d], F(lp.attB+2*e+1), ad[e], NNh[et.d], heads);
      hipMemsetAsync(me[e],  0, (size_t)NNh[et.d]*heads*4, stream);
      hipMemsetAsync(ss[e],  0, (size_t)NNh[et.d]*heads*4, stream);
      hipMemsetAsync(agg[e], 0, (size_t)NNh[et.d]*64*4,    stream);
      const long EH = (long)et.E * heads;
      edge_max<<<cdiv(EH,256),256,0,stream>>>(as[e], ad[e], I(et.si), I(et.di),
                                              (unsigned*)me[e], et.E, heads);
      edge_sum<<<cdiv(EH,256),256,0,stream>>>(as[e], ad[e], I(et.si), I(et.di),
                                              (const unsigned*)me[e], ss[e], et.E, heads);
      edge_scatter<<<cdiv(EH,256),256,0,stream>>>(H[et.s], as[e], ad[e], I(et.si), I(et.di),
                                                  (const unsigned*)me[e], ss[e], agg[e], et.E, heads);
    }

    for (int dt = 0; dt < 2; ++dt) {                // semantic attention per dest type
      int list[5], M = 0;
      for (int e = 0; e < 7; ++e) {
        if (!lp.ci && e == 4) continue;
        if (ets[e].d == dt) list[M++] = e;
      }
      const int Nd = NNh[dt];
      hipMemsetAsync(colsum, 0, (size_t)M * 64 * 4, stream);
      for (int m = 0; m < M; ++m) {
        const int e = list[m];
        relu_ip<<<cdiv((long)Nd*64,256),256,0,stream>>>(agg[e], (size_t)Nd*64);
        gemm64_bias<<<cdiv(Nd,128),256,0,stream>>>(agg[e], wbf[lp.kwi], F(lp.kbi), T, Nd);
        tanh_colsum<<<256,256,0,stream>>>(T, colsum + m * 64, Nd);
      }
      score_softmax<<<1,32,0,stream>>>(colsum, F(lp.qi), attnb, M, 1.f / (float)Nd);
      Ptr5 ps{}; for (int m = 0; m < M; ++m) ps.p[m] = agg[list[m]];
      combine<<<cdiv((long)Nd*64,256),256,0,stream>>>(ps, attnb, res[dt], M, (size_t)Nd*64);
      elu_ln<<<cdiv(Nd,256),256,0,stream>>>(res[dt],
              lp.ln ? F(lp.lng) : (const float*)nullptr,
              lp.ln ? F(lp.lnb) : (const float*)nullptr, Nd, lp.ln ? 1 : 0);
    }
  }

  // ----- set transformers (heads=4, d=16, P_IND=16) -----
  for (int t = 0; t < 2; ++t) {
    const int pb = (t == 0) ? 86 : 93;
    const int Nn = NNh[t];
    float* Kb = (t == 0) ? agg[0] : agg[2];         // reuse inst/var-sized agg buffers
    float* Vb = (t == 0) ? agg[1] : agg[6];
    const __bf16* inw = wbf[pb];
    const float*  inb = F(pb + 1);
    gemm64_bias<<<cdiv(Nn,128),256,0,stream>>>(res[t], inw + 64*64,  inb + 64,  Kb, Nn);
    gemm64_bias<<<cdiv(Nn,128),256,0,stream>>>(res[t], inw + 128*64, inb + 128, Vb, Nn);
    gemm64_bias<<<1,256,0,stream>>>(F(pb + 4), inw, inb, Qb, 16);   // q = ind @ Wq^T + bq
    st_attn<<<64,256,0,stream>>>(Qb, Kb, Vb, Ob, Nn);
    st_mean<<<1,64,0,stream>>>(Ob, F(pb + 2), F(pb + 3), om);
    small_linear<<<1,64,0,stream>>>(om, F(pb + 5), F(pb + 6), stout + 64 * t, 64, 64, 0);
  }

  // ----- head -----
  small_linear<<<1,64,0,stream>>>(stout,      F(100), F(101), gb,           64, 64,  1); // gelu
  small_linear<<<1,64,0,stream>>>(stout + 64, F(102), F(103), gb + 64,      64, 64,  1); // gelu
  small_linear<<<1,64,0,stream>>>(gb,         F(104), F(105), g2,           64, 128, 2); // leaky 0.01
  small_linear<<<1,64,0,stream>>>(g2,         F(106), F(107), (float*)d_out, 64, 64,  3); // relu
}
